// NeighborhoodSelfAttentionBlock_37460704756335
// MI455X (gfx1250) — matrix operand order repaired and checked by reading, not compiled, not run on hardware
//
#include <hip/hip_runtime.h>
#include <hip/hip_bf16.h>

typedef __attribute__((ext_vector_type(16))) __bf16        v16bf;
typedef __attribute__((ext_vector_type(8)))  float         v8f;
typedef __attribute__((ext_vector_type(8)))  unsigned int  v8u;

// Exact pointee type of the async-LDS builtin params (from the clang diagnostic):
// '__attribute__((__vector_size__(4 * sizeof(int)))) int __device__ *'
typedef int v4i_vs __attribute__((vector_size(4 * sizeof(int))));

// Use CDNA5 async global->LDS copies (ASYNCcnt) when the toolchain exposes them.
#if defined(__HIP_DEVICE_COMPILE__) && \
    __has_builtin(__builtin_amdgcn_global_load_async_to_lds_b128) && \
    __has_builtin(__builtin_amdgcn_s_wait_asynccnt)
#define USE_ASYNC_LDS 1
#else
#define USE_ASYNC_LDS 0
#endif

// ---------------- helpers ----------------
static __device__ __forceinline__ unsigned short f2bf(float f) {
    unsigned int u = __float_as_uint(f);
    u = (u + 0x7FFFu + ((u >> 16) & 1u)) >> 16;   // round-to-nearest-even
    return (unsigned short)u;
}

static __device__ __forceinline__ float wsum(float v) {
    #pragma unroll
    for (int m = 16; m > 0; m >>= 1) v += __shfl_xor(v, m, 32);
    return v;
}

union FragU { uint4 q[2]; v8u v; };

// ---------------- 1. f32 -> bf16 convert (weights keep [N][K] layout == B^T) --------
__global__ __launch_bounds__(256)
void cvt_kernel(const float* __restrict__ in, unsigned short* __restrict__ out, int n) {
    int i = blockIdx.x * 256 + threadIdx.x;
    if (i < n) out[i] = f2bf(in[i]);
}

// ---------------- 2. cond embedding: se = emb_w @ cond + emb_b -> sc, sh ------------
__global__ __launch_bounds__(256)
void condemb_kernel(const float* __restrict__ emb_w, const float* __restrict__ emb_b,
                    const float* __restrict__ cond, float* __restrict__ sc,
                    float* __restrict__ sh) {
    int gid  = blockIdx.x * 8 + (threadIdx.x >> 5);   // 0..4095  (b,o)
    int lane = threadIdx.x & 31;
    int b = gid >> 10, o = gid & 1023;
    const float* wr = emb_w + (size_t)o * 512;
    const float* cr = cond + b * 512;
    float acc = 0.f;
    for (int e = lane; e < 512; e += 32) acc += wr[e] * cr[e];
    acc = wsum(acc);
    if (lane == 0) {
        acc += emb_b[o];
        if (o < 512) sc[b * 512 + o] = acc;
        else         sh[b * 512 + (o - 512)] = acc;
    }
}

// ---------------- 3. GroupNorm stats: one block per (b, g) --------------------------
__global__ __launch_bounds__(256)
void gnstats_kernel(const float* __restrict__ x, float* __restrict__ stats) {
    int bg = blockIdx.x;                              // 0..127
    const float* base = x + ((size_t)bg << 14);       // 16 ch * 1024 hw contiguous
    float s = 0.f, s2 = 0.f;
    for (int i = threadIdx.x; i < 16384; i += 256) {
        float v = base[i]; s += v; s2 += v * v;
    }
    s = wsum(s); s2 = wsum(s2);
    __shared__ float red[16];
    int wv = threadIdx.x >> 5, lane = threadIdx.x & 31;
    if (lane == 0) { red[wv] = s; red[8 + wv] = s2; }
    __syncthreads();
    if (threadIdx.x == 0) {
        float ts = 0.f, ts2 = 0.f;
        #pragma unroll
        for (int i = 0; i < 8; ++i) { ts += red[i]; ts2 += red[8 + i]; }
        float mu  = ts * (1.f / 16384.f);
        float var = ts2 * (1.f / 16384.f) - mu * mu;
        stats[bg * 2]     = mu;
        stats[bg * 2 + 1] = rsqrtf(var + 1e-6f);
    }
}

// ------- 4. fused GN-apply + (1+sc)*x + sh, NCHW->[token][channel] bf16 -------------
__global__ __launch_bounds__(256)
void modulate_kernel(const float* __restrict__ x, const float* __restrict__ stats,
                     const float* __restrict__ gn_g, const float* __restrict__ gn_b,
                     const float* __restrict__ sc, const float* __restrict__ sh,
                     unsigned short* __restrict__ xn) {
    __shared__ unsigned short tile[32 * 33];
    int b      = blockIdx.x >> 4;
    int cbase  = (blockIdx.x & 15) * 32;
    int hwbase = blockIdx.y * 32;
    #pragma unroll
    for (int rep = 0; rep < 4; ++rep) {               // coalesced reads along hw
        int i = rep * 256 + threadIdx.x;
        int cl = i >> 5, wl = i & 31;
        int c  = cbase + cl;
        float v    = x[(((size_t)(b * 512 + c)) << 10) + hwbase + wl];
        float mu   = stats[(b * 32 + (c >> 4)) * 2];
        float rstd = stats[(b * 32 + (c >> 4)) * 2 + 1];
        v = (v - mu) * rstd * gn_g[c] + gn_b[c];
        v = v * (1.f + sc[b * 512 + c]) + sh[b * 512 + c];
        tile[cl * 33 + wl] = f2bf(v);
    }
    __syncthreads();
    #pragma unroll
    for (int rep = 0; rep < 4; ++rep) {               // coalesced writes along c
        int j = rep * 256 + threadIdx.x;
        int cl = j & 31, wl = j >> 5;
        xn[((size_t)(b * 1024 + hwbase + wl) << 9) + cbase + cl] = tile[cl * 33 + wl];
    }
}

// ---------------- 5/8. WMMA GEMM: C[M,N] = A[M,512] * W[N,512]^T --------------------
// Block tile 128x64, 8 waves, each wave 32x32 via 2x2 v_wmma_f32_16x16x32_bf16.
// Double-buffered LDS; stages filled by global_load_async_to_lds_b128 (ASYNCcnt)
// when available, so tile k+1 streams into LDS while tile k feeds the WMMAs.
// MODE 0: write f32 row-major [M][N] (QKV).  MODE 1: residual add, scatter to NCHW.
template <int MODE>
__global__ __launch_bounds__(256)
void gemm_kernel(const unsigned short* __restrict__ A,
                 const unsigned short* __restrict__ Wb,
                 float* __restrict__ outF, float* __restrict__ dOut,
                 const float* __restrict__ xres, int N) {
#if USE_ASYNC_LDS
    __shared__ unsigned short As[2][128 * 40];        // 80B rows (16B-aligned)
    __shared__ unsigned short Bs[2][64 * 40];
#else
    __shared__ unsigned short As[1][128 * 40];
    __shared__ unsigned short Bs[1][64 * 40];
#endif
    const int tid  = threadIdx.x;
    const int lane = tid & 31;
    const int wave = tid >> 5;
    const int wm   = wave & 3;                        // 4 M-subtiles
    const int wn   = wave >> 2;                       // 2 N-subtiles
    const int hf   = lane >> 4;
    const int l16  = lane & 15;
    const int m0   = blockIdx.x * 128;
    const int n0   = blockIdx.y * 64;

    const int aRow = tid >> 2, aCol = (tid & 3) * 8;  // 8-bf16 (16B) chunks
    const unsigned short* gA0 = A  + (size_t)(m0 + aRow)      * 512 + aCol;
    const unsigned short* gA1 = A  + (size_t)(m0 + aRow + 64) * 512 + aCol;
    const unsigned short* gB  = Wb + (size_t)(n0 + aRow)      * 512 + aCol;
    const int lOffA0 = aRow * 40 + aCol;
    const int lOffA1 = (aRow + 64) * 40 + aCol;

    v8f acc[2][2] = {};

#if USE_ASYNC_LDS
    // prologue: stream stage 0 into buffer 0, bypassing VGPRs
    __builtin_amdgcn_global_load_async_to_lds_b128(
        (v4i_vs*)gA0, (v4i_vs*)&As[0][lOffA0], 0, 0);
    __builtin_amdgcn_global_load_async_to_lds_b128(
        (v4i_vs*)gA1, (v4i_vs*)&As[0][lOffA1], 0, 0);
    __builtin_amdgcn_global_load_async_to_lds_b128(
        (v4i_vs*)gB,  (v4i_vs*)&Bs[0][lOffA0], 0, 0);
#endif

    for (int ks = 0; ks < 16; ++ks) {
#if USE_ASYNC_LDS
        const int buf = ks & 1;
        __syncthreads();                              // all waves done reading buf^1
        if (ks + 1 < 16) {
            const int kn = (ks + 1) * 32;
            __builtin_amdgcn_global_load_async_to_lds_b128(
                (v4i_vs*)(gA0 + kn), (v4i_vs*)&As[buf ^ 1][lOffA0], 0, 0);
            __builtin_amdgcn_global_load_async_to_lds_b128(
                (v4i_vs*)(gA1 + kn), (v4i_vs*)&As[buf ^ 1][lOffA1], 0, 0);
            __builtin_amdgcn_global_load_async_to_lds_b128(
                (v4i_vs*)(gB + kn),  (v4i_vs*)&Bs[buf ^ 1][lOffA0], 0, 0);
            __builtin_amdgcn_s_wait_asynccnt(3);      // in-order: stage ks landed
        } else {
            __builtin_amdgcn_s_wait_asynccnt(0);
        }
        __syncthreads();                              // buf data visible to all waves
#else
        const int buf = 0;
        const int k0  = ks * 32;
        *(uint4*)(&As[0][lOffA0]) = *(const uint4*)(gA0 + k0);
        *(uint4*)(&As[0][lOffA1]) = *(const uint4*)(gA1 + k0);
        *(uint4*)(&Bs[0][lOffA0]) = *(const uint4*)(gB + k0);
        if (k0 + 32 < 512) {                          // gfx1250 global_prefetch_b8
            __builtin_prefetch(gA0 + k0 + 32, 0, 3);
            __builtin_prefetch(gB  + k0 + 32, 0, 3);
        }
        __syncthreads();
#endif
        // A fragment: lane m=l16, half hf; e<8 -> k=e+8*hf ; e>=8 -> k=16+(e-8)+8*hf
        v8u araw[2], braw[2];
        #pragma unroll
        for (int s = 0; s < 2; ++s) {
            const unsigned short* ap = &As[buf][(wm * 32 + s * 16 + l16) * 40];
            FragU u;
            u.q[0] = *(const uint4*)(ap + 8 * hf);
            u.q[1] = *(const uint4*)(ap + 16 + 8 * hf);
            araw[s] = u.v;
        }
        // B fragment: lane n=l16, half hf; e<8 -> k=e+8*hf ; e>=8 -> k=16+(e-8)+8*hf
        #pragma unroll
        for (int t = 0; t < 2; ++t) {
            const unsigned short* bp = &Bs[buf][(wn * 32 + t * 16 + l16) * 40];
            FragU u;
            u.q[0] = *(const uint4*)(bp + 8 * hf);
            u.q[1] = *(const uint4*)(bp + 16 + 8 * hf);
            braw[t] = u.v;
        }
        #pragma unroll
        for (int s = 0; s < 2; ++s)
            #pragma unroll
            for (int t = 0; t < 2; ++t)
                acc[s][t] = __builtin_amdgcn_wmma_f32_16x16x32_bf16(
                    false, __builtin_bit_cast(v16bf, araw[s]),
                    false, __builtin_bit_cast(v16bf, braw[t]),
                    (short)0, acc[s][t], false, false);
#if !USE_ASYNC_LDS
        __syncthreads();
#endif
    }

    // epilogue: VGPR r -> M = 8*hf + r, N = l16
    #pragma unroll
    for (int s = 0; s < 2; ++s)
        #pragma unroll
        for (int t = 0; t < 2; ++t)
            #pragma unroll
            for (int r = 0; r < 8; ++r) {
                int m = m0 + wm * 32 + s * 16 + hf * 8 + r;
                int n = n0 + wn * 32 + t * 16 + l16;
                float v = acc[s][t][r];
                if (MODE == 0) {
                    outF[(size_t)m * N + n] = v;
                } else {
                    int b = m >> 10, hw = m & 1023;
                    size_t idx = ((size_t)(b * 512 + n) << 10) + hw;
                    dOut[idx] = v + xres[idx];        // residual, NCHW scatter
                }
            }
}

// ---------------- 6. per-head LayerNorm + 2D RoPE on q,k (in place) -----------------
__global__ __launch_bounds__(256)
void lnrope_kernel(float* __restrict__ qkv,
                   const float* __restrict__ qn_g, const float* __restrict__ qn_b,
                   const float* __restrict__ kn_g, const float* __restrict__ kn_b) {
    int gid  = blockIdx.x * 8 + (threadIdx.x >> 5);   // (token,head), 32768 waves
    int lane = threadIdx.x & 31;                      // pair index i: dims 2i, 2i+1
    int head = gid & 7;
    int p    = gid >> 3;
    int hpos = (p >> 5) & 31, wpos = p & 31;
    int d0 = lane * 2, d1 = d0 + 1;

    int   j    = lane & 15;
    float freq = __powf(10000.f, -(float)j * (1.f / 16.f));
    float pos  = (lane < 16) ? (float)hpos : (float)wpos;
    float cs   = __cosf(pos * freq);
    float sn   = __sinf(pos * freq);

    float* base = qkv + (size_t)p * 1536 + head * 64;
    #pragma unroll
    for (int t = 0; t < 2; ++t) {                     // t=0: q, t=1: k
        float* ptr      = base + t * 512;
        const float* g  = t ? kn_g : qn_g;
        const float* bb = t ? kn_b : qn_b;
        float v0 = ptr[d0], v1 = ptr[d1];
        float mu   = wsum(v0 + v1) * (1.f / 64.f);
        float ms   = wsum(v0 * v0 + v1 * v1) * (1.f / 64.f);
        float rstd = rsqrtf(ms - mu * mu + 1e-6f);
        v0 = (v0 - mu) * rstd * g[d0] + bb[d0];
        v1 = (v1 - mu) * rstd * g[d1] + bb[d1];
        ptr[d0] = v0 * cs - v1 * sn;                  // rot_half: (-x_odd, x_even)
        ptr[d1] = v1 * cs + v0 * sn;
    }
}

// ---------------- 7. 7x7 neighborhood attention, online softmax, wave/query ---------
__global__ __launch_bounds__(256)
void attn_kernel(const float* __restrict__ qkv, unsigned short* __restrict__ obf) {
    int gid  = blockIdx.x * 8 + (threadIdx.x >> 5);
    int lane = threadIdx.x & 31;
    int wpos = gid & 31, hpos = (gid >> 5) & 31;
    int head = (gid >> 10) & 7, b = gid >> 13;
    int p  = (b * 32 + hpos) * 32 + wpos;
    int d0 = lane * 2, d1 = d0 + 1;

    const float* qp = qkv + (size_t)p * 1536 + head * 64;
    float q0 = qp[d0], q1 = qp[d1];
    int hst = min(max(hpos - 3, 0), 25);
    int wst = min(max(wpos - 3, 0), 25);

    float mmax = -1e30f, ssum = 0.f, o0 = 0.f, o1 = 0.f;
    for (int kh = 0; kh < 7; ++kh)
        for (int kw = 0; kw < 7; ++kw) {
            int kp = (b * 32 + hst + kh) * 32 + wst + kw;
            const float* kb = qkv + (size_t)kp * 1536 + 512 + head * 64;
            float l = wsum(q0 * kb[d0] + q1 * kb[d1]) * 0.125f;  // all-lane reduce
            const float* vb = kb + 512;
            float vv0 = vb[d0], vv1 = vb[d1];
            if (l > mmax) {                            // uniform across wave
                float sc = __expf(mmax - l);
                ssum *= sc; o0 *= sc; o1 *= sc;
                mmax = l;
            }
            float pw = __expf(l - mmax);
            ssum += pw; o0 += pw * vv0; o1 += pw * vv1;
        }
    float inv = 1.f / ssum;
    unsigned short* op = obf + (size_t)p * 512 + head * 64;
    op[d0] = f2bf(o0 * inv);
    op[d1] = f2bf(o1 * inv);
}

// ---------------- launch ----------------
extern "C" void kernel_launch(void* const* d_in, const int* in_sizes, int n_in,
                              void* d_out, int out_size, void* d_ws, size_t ws_size,
                              hipStream_t stream) {
    const float* x     = (const float*)d_in[0];
    const float* cond  = (const float*)d_in[1];
    const float* emb_w = (const float*)d_in[2];
    const float* emb_b = (const float*)d_in[3];
    const float* gn_g  = (const float*)d_in[4];
    const float* gn_b  = (const float*)d_in[5];
    const float* qkv_w = (const float*)d_in[6];
    const float* out_w = (const float*)d_in[7];
    const float* qn_g  = (const float*)d_in[8];
    const float* qn_b  = (const float*)d_in[9];
    const float* kn_g  = (const float*)d_in[10];
    const float* kn_b  = (const float*)d_in[11];
    float* out = (float*)d_out;

    char* ws = (char*)d_ws;
    size_t off = 0;
    auto alloc = [&](size_t bytes) {
        void* p = ws + off;
        off = (off + bytes + 255) & ~(size_t)255;
        return p;
    };
    float*          scb   = (float*)alloc(4 * 512 * 4);
    float*          shb   = (float*)alloc(4 * 512 * 4);
    float*          stats = (float*)alloc(128 * 2 * 4);
    unsigned short* xn    = (unsigned short*)alloc((size_t)4096 * 512 * 2);
    unsigned short* wqkv  = (unsigned short*)alloc((size_t)1536 * 512 * 2);
    unsigned short* wout  = (unsigned short*)alloc((size_t)512 * 512 * 2);
    float*          qkvf  = (float*)alloc((size_t)4096 * 1536 * 4);
    unsigned short* obf   = (unsigned short*)alloc((size_t)4096 * 512 * 2);

    cvt_kernel<<<3072, 256, 0, stream>>>(qkv_w, wqkv, 1536 * 512);
    cvt_kernel<<<1024, 256, 0, stream>>>(out_w, wout, 512 * 512);
    condemb_kernel<<<512, 256, 0, stream>>>(emb_w, emb_b, cond, scb, shb);
    gnstats_kernel<<<128, 256, 0, stream>>>(x, stats);
    modulate_kernel<<<dim3(64, 32), 256, 0, stream>>>(x, stats, gn_g, gn_b, scb, shb, xn);
    gemm_kernel<0><<<dim3(32, 24), 256, 0, stream>>>(xn, wqkv, qkvf, nullptr, nullptr, 1536);
    lnrope_kernel<<<4096, 256, 0, stream>>>(qkvf, qn_g, qn_b, kn_g, kn_b);
    attn_kernel<<<4096, 256, 0, stream>>>(qkvf, obf);
    gemm_kernel<1><<<dim3(32, 8), 256, 0, stream>>>(obf, wout, nullptr, out, x, 512);
}
